// MyModel_11879879544011
// MI455X (gfx1250) — matrix operand order repaired and checked by reading, not compiled, not run on hardware
//
#include <hip/hip_runtime.h>
#include <hip/hip_bf16.h>

// ---------------------------------------------------------------------------
// 12-layer LSTM, B=32, T=256, F=U=512.  bf16 WMMA (V_WMMA_F32_16X16X32_BF16).
// Per layer: batched input GEMM (all timesteps, fully parallel) + ONE
// persistent recurrent kernel that loops all 256 timesteps internally with a
// global (inter-block) barrier per step.  Cell state lives in registers.
// ---------------------------------------------------------------------------

typedef __attribute__((ext_vector_type(16))) __bf16 v16bf;
typedef __attribute__((ext_vector_type(8)))  __bf16 v8bf;
typedef __attribute__((ext_vector_type(8)))  float  v8f;

#define L_   12
#define U_   512
#define B_   32
#define T_   256
#define F_   512
#define NG_  2048     // 4*U
#define KB_  16       // K=512 / 32 per WMMA
#define NTU_ 32       // 512/16 N-tiles per gate
#define NTALL_ 128    // 2048/16 N-tiles total
#define KSPLIT_ 4     // waves cooperating on one recurrent tile
#define NB_REC_ 64    // recurrent blocks = 2 m-tiles * 32 u-tiles

// ---------------- weight pre-pack into WMMA B-fragment layout --------------
// B (32x16 KxN bf16): n = lane&15, K = kb*32 + 8*(lane>>4) + (e&7) + 16*(e>>3)
// Packed: ((((l*128 + nt)*16 + kb)*32 + lane)*16 + e)
__global__ void pack_weights_k(const float* __restrict__ src, __bf16* __restrict__ dst,
                               int total) {
    int idx = blockIdx.x * blockDim.x + threadIdx.x;
    if (idx >= total) return;
    int e    = idx & 15;
    int lane = (idx >> 4) & 31;
    int kb   = (idx >> 9) & 15;
    int nt   = (idx >> 13) & 127;
    int l    = idx >> 20;
    int half = lane >> 4;
    int n    = nt * 16 + (lane & 15);
    int k    = kb * 32 + 8 * half + (e & 7) + 16 * (e >> 3);
    dst[idx] = (__bf16)src[(size_t)l * (F_ * NG_) + (size_t)k * NG_ + n];
}

__global__ void zero_u32_k(unsigned* __restrict__ p, int n) {
    int i = blockIdx.x * blockDim.x + threadIdx.x;
    if (i < n) p[i] = 0u;
}

// ---------------- A-fragment builders (16-bit A 16x32 layout) --------------
// element e<8 : K = 8*half + e ; e>=8 : K = 16 + 8*half + (e-8)
__device__ __forceinline__ v16bf load_a_from_f32(const float* __restrict__ p, int half) {
    const float* p0 = p + 8 * half;
    const float* p1 = p + 16 + 8 * half;
    v16bf a;
#pragma unroll
    for (int j = 0; j < 8; ++j) a[j] = (__bf16)p0[j];
#pragma unroll
    for (int j = 0; j < 8; ++j) a[8 + j] = (__bf16)p1[j];
    return a;
}
__device__ __forceinline__ v16bf load_a_from_bf16(const __bf16* __restrict__ p, int half) {
    union { v16bf v; v8bf h[2]; } u;
    u.h[0] = *(const v8bf*)(p + 8 * half);
    u.h[1] = *(const v8bf*)(p + 16 + 8 * half);
    return u.v;
}

struct Frag4 { v16bf b[4]; };
__device__ __forceinline__ Frag4 load_bfrag4(const __bf16* __restrict__ pack,
                                             int ut, int kb, int lane) {
    Frag4 f;
#pragma unroll
    for (int g = 0; g < 4; ++g)
        f.b[g] = *(const v16bf*)(pack + ((size_t)(g * NTU_ + ut) * KB_ + kb) * 512 + lane * 16);
    return f;
}

__device__ __forceinline__ float sigmoid_f(float x) {
    return 1.0f / (1.0f + __expf(-x));
}

// ---------------- batched input GEMM: Zin = inp @ W + bias ------------------
// M = T*B = 8192 (row m = t*32 + b), N = 2048, K = 512.
// One wave per (mt, ut): 4 gate accumulators sharing one A-fragment.
// Double-buffered over kb; branch-free via LAYER0 template.
template <int LAYER0>
__global__ __launch_bounds__(256)
void lstm_input_gemm(const float*  __restrict__ x,     // (B,T,F), LAYER0 only
                     const __bf16* __restrict__ hin,   // (T,B,U), otherwise
                     const __bf16* __restrict__ wpack, // this layer, packed
                     const float*  __restrict__ bias,  // this layer (2048)
                     float*        __restrict__ zin) { // (T*B, 2048)
    int wave = blockIdx.x * (blockDim.x >> 5) + (threadIdx.x >> 5);
    int lane = threadIdx.x & 31;
    int mt   = wave >> 5;          // 0..511
    int ut   = wave & 31;          // 0..31
    int half = lane >> 4;
    int nl   = lane & 15;

    int m = mt * 16 + nl;          // A-fragment row: global M = t*B + b
    int t = m >> 5, b = m & 31;

    v8f acc[4];
#pragma unroll
    for (int g = 0; g < 4; ++g) {
        float bv = bias[g * U_ + ut * 16 + nl];
#pragma unroll
        for (int r = 0; r < 8; ++r) acc[g][r] = bv;
    }

    const float*  xrow = LAYER0 ? (x + ((size_t)b * T_ + t) * F_) : (const float*)nullptr;
    const __bf16* hrow = LAYER0 ? (const __bf16*)nullptr : (hin + ((size_t)t * B_ + b) * U_);

    v16bf a  = LAYER0 ? load_a_from_f32(xrow, half) : load_a_from_bf16(hrow, half);
    Frag4 bf = load_bfrag4(wpack, ut, 0, lane);

#pragma unroll 4
    for (int kb = 0; kb < KB_; ++kb) {
        v16bf a_n;
        Frag4 bf_n;
        if (kb + 1 < KB_) {   // prefetch next fragments before issuing WMMAs
            a_n  = LAYER0 ? load_a_from_f32(xrow + (kb + 1) * 32, half)
                          : load_a_from_bf16(hrow + (kb + 1) * 32, half);
            bf_n = load_bfrag4(wpack, ut, kb + 1, lane);
        }
#pragma unroll
        for (int g = 0; g < 4; ++g)
            acc[g] = __builtin_amdgcn_wmma_f32_16x16x32_bf16(
                         false, a, false, bf.b[g], (short)0, acc[g], false, false);
        a = a_n; bf = bf_n;
    }

    // C/D layout: m = 8*half + r, n = lane&15
#pragma unroll
    for (int g = 0; g < 4; ++g) {
        int col = g * U_ + ut * 16 + nl;
#pragma unroll
        for (int r = 0; r < 8; ++r) {
            int mc = mt * 16 + 8 * half + r;
            zin[(size_t)mc * NG_ + col] = acc[g][r];
        }
    }
}

// ---------------- inter-block barrier (counter per timestep) ---------------
__device__ __forceinline__ void grid_barrier(unsigned* __restrict__ ctr, unsigned nb) {
    __threadfence();
    __syncthreads();
    if (threadIdx.x == 0) {
        unsigned cnt = atomicAdd(ctr, 1u) + 1u;
        while (cnt < nb) {
            __builtin_amdgcn_s_sleep(2);
            cnt = atomicAdd(ctr, 0u);
        }
    }
    __syncthreads();
    __threadfence();
}

// ---------------- persistent recurrent kernel (one per layer) ---------------
// 64 blocks, one (mt, ut) output tile each; 4 waves split K (4 kb each).
// LDS reduction of partial accumulators; wave 0 runs the fused gate math.
// Cell state c stays in wave-0 registers for the whole sequence.
__global__ __launch_bounds__(128)
void lstm_recurrent_layer(const float*  __restrict__ zin,   // (T*B, 2048)
                          const __bf16* __restrict__ rpack, // packed R, this layer
                          __bf16*       __restrict__ hself, // (T,B,U) this layer out
                          float*        __restrict__ hf32,  // (B,U), written at t=T-1
                          unsigned*     __restrict__ bctr) {// T_ counters (pre-zeroed)
    __shared__ float red[(KSPLIT_ - 1) * 4 * 8 * 32];       // 12 KB

    int lane = threadIdx.x & 31;
    int ks   = threadIdx.x >> 5;   // 0..3 K-split id
    int mt   = (int)blockIdx.x >> 5;      // 0..1
    int ut   = (int)blockIdx.x & 31;      // 0..31
    int half = lane >> 4;
    int nl   = lane & 15;

    float creg[8];
#pragma unroll
    for (int r = 0; r < 8; ++r) creg[r] = 0.0f;

    for (int t = 0; t < T_; ++t) {
        v8f acc[4];
        if (ks == 0) {
#pragma unroll
            for (int g = 0; g < 4; ++g)
#pragma unroll
                for (int r = 0; r < 8; ++r) {
                    int m = mt * 16 + 8 * half + r;
                    acc[g][r] = zin[((size_t)t * B_ + m) * NG_ + g * U_ + ut * 16 + nl];
                }
        } else {
#pragma unroll
            for (int g = 0; g < 4; ++g)
#pragma unroll
                for (int r = 0; r < 8; ++r) acc[g][r] = 0.0f;
        }

        if (t > 0) {
            const __bf16* hrow = hself + ((size_t)(t - 1) * B_ + (mt * 16 + nl)) * U_;
            int kb0 = ks * 4;
            v16bf a  = load_a_from_bf16(hrow + kb0 * 32, half);
            Frag4 bf = load_bfrag4(rpack, ut, kb0, lane);
#pragma unroll
            for (int j = 0; j < 4; ++j) {
                v16bf a_n;
                Frag4 bf_n;
                if (j + 1 < 4) {
                    a_n  = load_a_from_bf16(hrow + (kb0 + j + 1) * 32, half);
                    bf_n = load_bfrag4(rpack, ut, kb0 + j + 1, lane);
                }
#pragma unroll
                for (int g = 0; g < 4; ++g)
                    acc[g] = __builtin_amdgcn_wmma_f32_16x16x32_bf16(
                                 false, a, false, bf.b[g], (short)0, acc[g], false, false);
                a = a_n; bf = bf_n;
            }

            if (ks > 0) {
#pragma unroll
                for (int g = 0; g < 4; ++g)
#pragma unroll
                    for (int r = 0; r < 8; ++r)
                        red[((ks - 1) * 4 + g) * 256 + r * 32 + lane] = acc[g][r];
            }
            __syncthreads();
            if (ks == 0) {
#pragma unroll
                for (int w = 1; w < KSPLIT_; ++w)
#pragma unroll
                    for (int g = 0; g < 4; ++g)
#pragma unroll
                        for (int r = 0; r < 8; ++r)
                            acc[g][r] += red[((w - 1) * 4 + g) * 256 + r * 32 + lane];
            }
        }

        if (ks == 0) {
#pragma unroll
            for (int r = 0; r < 8; ++r) {
                float ig = sigmoid_f(acc[0][r]);
                float fg = sigmoid_f(acc[1][r]);
                float gg = tanhf(acc[2][r]);
                float og = sigmoid_f(acc[3][r]);
                float cn = fg * creg[r] + ig * gg;
                creg[r] = cn;
                float hn = og * tanhf(cn);
                int m = mt * 16 + 8 * half + r;
                int u = ut * 16 + nl;
                hself[((size_t)t * B_ + m) * U_ + u] = (__bf16)hn;
                if (t == T_ - 1) hf32[(size_t)m * U_ + u] = hn;
            }
        }

        grid_barrier(bctr + t, NB_REC_);
    }
}

__global__ void copy_out_k(const float* __restrict__ src, float* __restrict__ dst, int n) {
    int i = blockIdx.x * blockDim.x + threadIdx.x;
    if (i < n) dst[i] = src[i];
}

// ---------------------------------------------------------------------------
extern "C" void kernel_launch(void* const* d_in, const int* in_sizes, int n_in,
                              void* d_out, int out_size, void* d_ws, size_t ws_size,
                              hipStream_t stream) {
    const float* x    = (const float*)d_in[0];   // (B,T,F)
    const float* ker  = (const float*)d_in[1];   // (L,F,4U)
    const float* rec  = (const float*)d_in[2];   // (L,U,4U)
    const float* bias = (const float*)d_in[3];   // (L,4U)

    // Workspace layout (~128.1 MiB total)
    char* ws = (char*)d_ws;
    const size_t packElems = (size_t)L_ * NTALL_ * KB_ * 512;   // 12.58M per matrix set
    __bf16*   wpack = (__bf16*)ws;  ws += packElems * sizeof(__bf16);              // 24 MiB
    __bf16*   rpack = (__bf16*)ws;  ws += packElems * sizeof(__bf16);              // 24 MiB
    float*    zin   = (float*) ws;  ws += (size_t)T_ * B_ * NG_ * sizeof(float);   // 64 MiB
    __bf16*   hbuf0 = (__bf16*)ws;  ws += (size_t)T_ * B_ * U_ * sizeof(__bf16);   // 8 MiB
    __bf16*   hbuf1 = (__bf16*)ws;  ws += (size_t)T_ * B_ * U_ * sizeof(__bf16);   // 8 MiB
    float*    hf32  = (float*) ws;  ws += (size_t)B_ * U_ * sizeof(float);         // 64 KiB
    unsigned* bctr  = (unsigned*)ws; ws += (size_t)L_ * T_ * sizeof(unsigned);     // 12 KiB

    // Re-zero barrier counters every call (graph replays stay deterministic)
    zero_u32_k<<<(L_ * T_ + 255) / 256, 256, 0, stream>>>(bctr, L_ * T_);

    // One-time bf16 conversion + fragment packing of all weights
    {
        int total = (int)packElems;
        int thr = 256, blk = (total + thr - 1) / thr;
        pack_weights_k<<<blk, thr, 0, stream>>>(ker, wpack, total);
        pack_weights_k<<<blk, thr, 0, stream>>>(rec, rpack, total);
    }

    for (int l = 0; l < L_; ++l) {
        const __bf16* hin  = (l & 1) ? hbuf0 : hbuf1;  // previous layer's outputs
        __bf16*       hout = (l & 1) ? hbuf1 : hbuf0;  // this layer's outputs
        const __bf16* wp = wpack + (size_t)l * NTALL_ * KB_ * 512;
        const __bf16* rp = rpack + (size_t)l * NTALL_ * KB_ * 512;
        const float*  bl = bias + (size_t)l * NG_;

        // Fully parallel batched input GEMM over all timesteps (16384 waves)
        if (l == 0)
            lstm_input_gemm<1><<<2048, 256, 0, stream>>>(x, nullptr, wp, bl, zin);
        else
            lstm_input_gemm<0><<<2048, 256, 0, stream>>>(nullptr, hin, wp, bl, zin);

        // Persistent recurrence: 64 co-resident blocks, grid barrier per step
        lstm_recurrent_layer<<<NB_REC_, 128, 0, stream>>>(
            zin, rp, hout, hf32, bctr + (size_t)l * T_);
    }

    // h of top layer at t = T-1
    copy_out_k<<<(B_ * U_ + 255) / 256, 256, 0, stream>>>(hf32, (float*)d_out, B_ * U_);
}
